// HierarchicalAttention_25494925869655
// MI455X (gfx1250) — compile-verified, matrix operand order checked by reading, not run on hardware
//
#include <hip/hip_runtime.h>

// ---------------------------------------------------------------------------
// HierarchicalAttention fused kernel for gfx1250 (MI455X), bf16 WMMA pipeline.
// One workgroup (256 threads / 8 wave32) per batch element b in [0,2048).
// Phase 2 processes two heads concurrently (wave-group 0: even heads,
// wave-group 1: odd heads) to halve barrier count and double utilization.
// ---------------------------------------------------------------------------

typedef __bf16 bf16_t;
typedef bf16_t bf16x8  __attribute__((ext_vector_type(8)));
typedef bf16_t bf16x16 __attribute__((ext_vector_type(16)));
typedef float  f32x8   __attribute__((ext_vector_type(8)));

#define THREADS 256

// ---- LDS layout (byte offsets), regions overlaid across phases ----
// OFF_QB  : qb/kb/vb bf16 [3][32][128]               (lives whole kernel)
// OFF_MISC: biases / ln params f32                   (lives whole kernel)
// OFF_X   : phase1: xT bf16 [3][140][64]; phase2: head scratch group 0;
//           end of phase2: pacc f32 [16][256] partial-out2 combine
// OFF_WC  : phase1: conv weights bf16 [3][13][32][64]; phase2: head scratch
//           group 1 (first 47104 B); phase3: Wf bf16 [13][64][32] (0..53248),
//           out2T bf16 [140][32] (+53248), zbuf f32 [64][128] (+62208)
#define OFF_QB   0
#define OFF_MISC 24576
#define OFF_X    26752
#define OFF_WC   80512
#define SMEM_BYTES 240256

__device__ __forceinline__ bf16_t f2bf(float x) {
  union { float f; unsigned u; } un; un.f = x;
  unsigned r = un.u + 0x7FFFu + ((un.u >> 16) & 1u);   // round-to-nearest-even
  unsigned short h = (unsigned short)(r >> 16);
  return __builtin_bit_cast(bf16_t, h);
}

// Load a 16x32 bf16 WMMA fragment (A from [M][K] LDS, B from [N][K] LDS).
// ISA 16-bit A layout: lanes 0-15 hold K 0..7 (V0-3) and 16..23 (V4-7);
// lanes 16-31 hold K 8..15 and 24..31. Two ds_load_b128 per lane.
__device__ __forceinline__ bf16x16 load_frag(const bf16_t* base, int row0, int ld, int k0) {
  const int lane = threadIdx.x & 31;
  const int half = lane >> 4, mr = lane & 15;
  const bf16_t* p = base + (row0 + mr) * ld + k0 + half * 8;
  bf16x8 lo = *(const bf16x8*)(p);
  bf16x8 hi = *(const bf16x8*)(p + 16);
  return __builtin_shufflevector(lo, hi, 0,1,2,3,4,5,6,7,8,9,10,11,12,13,14,15);
}

__device__ __forceinline__ f32x8 wmma_bf16(bf16x16 a, bf16x16 b, f32x8 c) {
  return __builtin_amdgcn_wmma_f32_16x16x32_bf16(false, a, false, b, (short)0, c, false, false);
}

__global__ void __launch_bounds__(THREADS, 1)
ha_kernel(const float* __restrict__ q, const float* __restrict__ k, const float* __restrict__ v,
          const float* __restrict__ cq_w, const float* __restrict__ cq_b,
          const float* __restrict__ ck_w, const float* __restrict__ ck_b,
          const float* __restrict__ cv_w, const float* __restrict__ cv_b,
          const float* __restrict__ w_qs, const float* __restrict__ w_ks, const float* __restrict__ w_vs,
          const float* __restrict__ proj_w, const float* __restrict__ proj_b,
          const float* __restrict__ fc_w, const float* __restrict__ fc_b,
          const float* __restrict__ ln_a, const float* __restrict__ ln_b,
          float* __restrict__ out)
{
  extern __shared__ char smem[];
  bf16_t* qkvb  = (bf16_t*)(smem + OFF_QB);     // [3][32][128] conv outputs (t-major)
  float*  misc  = (float*)(smem + OFF_MISC);
  float*  cbias = misc;                         // [3][32]
  float*  pbias = misc + 96;                    // [128]
  float*  fcb   = misc + 224;                   // [64]
  float*  lna   = misc + 288;                   // [128]
  float*  lnb   = misc + 416;                   // [128]
  bf16_t* xT    = (bf16_t*)(smem + OFF_X);      // phase1: [3][140][64] (row = l+6, c contiguous)
  bf16_t* wconv = (bf16_t*)(smem + OFF_WC);     // phase1: [3][13][32][64] ([t][c] per shift)

  const int tid  = threadIdx.x;
  const int wave = tid >> 5, lane = tid & 31;
  const int half = lane >> 4, col = lane & 15;
  const int grp  = tid >> 7;                    // wave-group (0: waves 0-3, 1: waves 4-7)
  const int gtid = tid & 127;                   // thread id within group
  const int wg   = (tid >> 5) & 3;              // wave id within group
  const int b    = blockIdx.x;

  const float* xin[3] = { q + (size_t)b * 8192, k + (size_t)b * 8192, v + (size_t)b * 8192 };
  const float* wc[3]  = { cq_w, ck_w, cv_w };
  const float* wh[3]  = { w_qs, w_ks, w_vs };

  // ---- small constants into LDS ----
  if (tid < 96)  cbias[tid] = (tid < 32) ? cq_b[tid] : (tid < 64 ? ck_b[tid - 32] : cv_b[tid - 64]);
  if (tid < 128) pbias[tid] = proj_b[tid];
  if (tid < 64)  fcb[tid]   = fc_b[tid];
  if (tid < 128) { lna[tid] = ln_a[tid]; lnb[tid] = ln_b[tid]; }

  // ---- phase 1 loads: transpose x to [l+6][c] bf16, conv weights to [j][t][c] bf16 ----
  for (int cvt = 0; cvt < 3; ++cvt) {
    const float* src = xin[cvt];
    bf16_t* dst = xT + cvt * 8960;              // 140*64
    for (int idx = tid; idx < 8192; idx += THREADS) {
      int c = idx >> 7, l = idx & 127;          // global (c,l), l contiguous -> coalesced
      dst[(l + 6) * 64 + c] = f2bf(src[idx]);
    }
    const float* ws = wc[cvt];
    bf16_t* wd = wconv + cvt * 26624;           // 13*32*64
    for (int idx = tid; idx < 26624; idx += THREADS) {
      int t = idx / 832, r = idx % 832, c = r / 13, j = r % 13;   // src (t,c,j)
      wd[(j * 32 + t) * 64 + c] = f2bf(ws[idx]);
    }
  }
  for (int idx = tid; idx < 3 * 12 * 64; idx += THREADS) {        // zero halo rows
    int cvt = idx / 768, r = idx % 768, rr = r >> 6, c = r & 63;
    int row = (rr < 6) ? rr : (128 + rr);       // rows 0..5 and 134..139
    xT[cvt * 8960 + row * 64 + c] = f2bf(0.f);
  }
  __syncthreads();

  // ---- phase 1 compute: 3 convs as 13 shifted GEMMs (M=32 t, N=128 l, K=64 c) ----
  // 48 output tiles of 16x16, 6 per wave; 26 WMMA per tile, software-pipelined.
  for (int i = 0; i < 6; ++i) {
    int tile = wave * 6 + i;
    int cvt = tile >> 4, rem = tile & 15;
    int mt = rem >> 3, nt = rem & 7;
    const bf16_t* W = wconv + cvt * 26624;
    const bf16_t* X = xT + cvt * 8960;
    f32x8 acc = {};
    bf16x16 fa = load_frag(W, mt * 16, 64, 0);
    bf16x16 fb = load_frag(X, nt * 16, 64, 0);
#pragma unroll
    for (int it = 0; it < 26; ++it) {
      int nx = (it + 1 < 26) ? it + 1 : it;     // clamp (redundant last reload)
      int j = nx >> 1, kc = nx & 1;
      bf16x16 na = load_frag(W + j * 2048, mt * 16, 64, kc * 32);
      bf16x16 nb = load_frag(X, nt * 16 + j, 64, kc * 32);
      acc = wmma_bf16(fa, fb, acc);
      fa = na; fb = nb;
    }
    bf16_t* dst = qkvb + cvt * 4096;
    const float* bias = cbias + cvt * 32;
#pragma unroll
    for (int r = 0; r < 8; ++r) {
      int t = mt * 16 + r + half * 8;           // C/D layout: M = r + 8*half, N = col
      dst[t * 128 + nt * 16 + col] = f2bf(acc[r] + bias[t]);
    }
  }
  __syncthreads();

  // ---- phase 2: two heads per iteration (group 0: even, group 1: odd) ----
  char*   hbase  = smem + (grp ? OFF_WC : OFF_X);   // 47104 B per-group scratch
  bf16_t* wT     = (bf16_t*)hbase;            // [3][32][128] w^T ([d][l])
  bf16_t* projW  = (bf16_t*)hbase + 12288;    // [128][32] proj slice ([m][d])
  bf16_t* qh_    = (bf16_t*)hbase + 16384;    // [32][32] [t][d]
  bf16_t* kh_    = (bf16_t*)hbase + 17408;    // [32][32] [t][d]
  bf16_t* vhT    = (bf16_t*)hbase + 18432;    // [32][32] [d][t]
  bf16_t* probs  = (bf16_t*)hbase + 19456;    // [32][32] [tq][tk]
  bf16_t* outh   = (bf16_t*)hbase + 20480;    // [32][32] [t][d]
  float*  scoresF = (float*)(hbase + 43008);  // [32][32] f32

  f32x8 acc2[2][2] = {};    // partial out2 tiles: (mt, nt = wg*2+nn), this group's heads
  for (int hh = 0; hh < 8; ++hh) {
    int h = hh * 2 + grp;
    // group-local loads: w_*[h] transposed to [d][l], proj slice [m][d]
    for (int m3 = 0; m3 < 3; ++m3) {
      const float* src = wh[m3] + (size_t)h * 4096;
      bf16_t* dst = wT + m3 * 4096;
      for (int idx = gtid; idx < 4096; idx += 128) {
        int l = idx >> 5, d = idx & 31;
        dst[d * 128 + l] = f2bf(src[idx]);
      }
    }
    for (int idx = gtid; idx < 4096; idx += 128)
      projW[idx] = f2bf(proj_w[(idx >> 5) * 512 + h * 32 + (idx & 31)]);
    // prefetch next head's weights into cache (global_prefetch_b8)
    int hn = (hh + 1) * 2 + grp;
    if (hn < 16) {
      __builtin_prefetch(w_qs + (size_t)hn * 4096 + gtid * 32);
      __builtin_prefetch(w_ks + (size_t)hn * 4096 + gtid * 32);
      __builtin_prefetch(w_vs + (size_t)hn * 4096 + gtid * 32);
      __builtin_prefetch(proj_w + (size_t)gtid * 512 + hn * 32);
    }
    __syncthreads();

    // qh/kh/vh: (32x128)*(128x32), 12 tiles per group, 3 per wave, K pipelined
    for (int i = 0; i < 3; ++i) {
      int tile = wg * 3 + i;                  // 0..11
      int m3 = tile >> 2, mt = (tile >> 1) & 1, ntq = tile & 1;
      const bf16_t* Ab = qkvb + m3 * 4096;
      const bf16_t* Bb = wT + m3 * 4096;
      f32x8 acc = {};
      bf16x16 fa = load_frag(Ab, mt * 16, 128, 0);
      bf16x16 fb = load_frag(Bb, ntq * 16, 128, 0);
#pragma unroll
      for (int kc = 0; kc < 4; ++kc) {
        int nx = (kc + 1 < 4) ? kc + 1 : kc;
        bf16x16 na = load_frag(Ab, mt * 16, 128, nx * 32);
        bf16x16 nb = load_frag(Bb, ntq * 16, 128, nx * 32);
        acc = wmma_bf16(fa, fb, acc);
        fa = na; fb = nb;
      }
#pragma unroll
      for (int r = 0; r < 8; ++r) {
        int t = mt * 16 + r + half * 8, d = ntq * 16 + col;
        if (m3 == 0)      qh_[t * 32 + d] = f2bf(acc[r]);
        else if (m3 == 1) kh_[t * 32 + d] = f2bf(acc[r]);
        else              vhT[d * 32 + t] = f2bf(acc[r]);   // transposed for attn*V
      }
    }
    __syncthreads();

    // scores = qh*kh^T (K=d=32): 4 tiles per group, 1 per wave
    {
      int mt = wg >> 1, ntq = wg & 1;
      f32x8 acc = {};
      acc = wmma_bf16(load_frag(qh_, mt * 16, 32, 0), load_frag(kh_, ntq * 16, 32, 0), acc);
#pragma unroll
      for (int r = 0; r < 8; ++r)
        scoresF[(mt * 16 + r + half * 8) * 32 + ntq * 16 + col] = acc[r];
    }
    __syncthreads();

    // softmax over tk (32) per row; 32 lanes per group
    if (gtid < 32) {
      const float scale = 0.08838834764831845f;   // 1/sqrt(128)
      float rowv[32];
      float mx = -1e30f;
#pragma unroll
      for (int j2 = 0; j2 < 32; ++j2) { rowv[j2] = scoresF[gtid * 32 + j2] * scale; mx = fmaxf(mx, rowv[j2]); }
      float s = 0.f;
#pragma unroll
      for (int j2 = 0; j2 < 32; ++j2) { rowv[j2] = __expf(rowv[j2] - mx); s += rowv[j2]; }
      float inv = 1.f / s;
#pragma unroll
      for (int j2 = 0; j2 < 32; ++j2) probs[gtid * 32 + j2] = f2bf(rowv[j2] * inv);
    }
    __syncthreads();

    // out_h = probs*vh (K=tk=32): 1 tile per wave
    {
      int mt = wg >> 1, ntq = wg & 1;
      f32x8 acc = {};
      acc = wmma_bf16(load_frag(probs, mt * 16, 32, 0), load_frag(vhT, ntq * 16, 32, 0), acc);
#pragma unroll
      for (int r = 0; r < 8; ++r)
        outh[(mt * 16 + r + half * 8) * 32 + ntq * 16 + col] = f2bf(acc[r]);
    }
    __syncthreads();

    // out2 partial += out_h * projW_h^T (M=t 32, N=m 128, K=d 32); wave owns nt = wg*2+nn
#pragma unroll
    for (int mt = 0; mt < 2; ++mt)
#pragma unroll
      for (int nn = 0; nn < 2; ++nn)
        acc2[mt][nn] = wmma_bf16(load_frag(outh, mt * 16, 32, 0),
                                 load_frag(projW, (wg * 2 + nn) * 16, 32, 0), acc2[mt][nn]);
    __syncthreads();   // protect per-group LDS regions for next iteration
  }

  // ---- combine even/odd head partials, add proj bias, build out2T ----
  float*  pacc  = (float*)(smem + OFF_X);              // [16 tiles][256] f32
  bf16_t* wf    = wconv;                               // [13][64][32] ([c][t] per shift)
  bf16_t* out2T = (bf16_t*)(smem + OFF_WC + 53248);    // [140][32] (row = m+6, t contiguous)
  float*  zbuf  = (float*)(smem + OFF_WC + 62208);     // [64][128] f32

  if (grp == 1) {
#pragma unroll
    for (int mt = 0; mt < 2; ++mt)
#pragma unroll
      for (int nn = 0; nn < 2; ++nn) {
        int tIdx = mt * 8 + wg * 2 + nn;
#pragma unroll
        for (int r = 0; r < 8; ++r)
          pacc[tIdx * 256 + (r + half * 8) * 16 + col] = acc2[mt][nn][r];
      }
  }
  __syncthreads();
  if (grp == 0) {
#pragma unroll
    for (int mt = 0; mt < 2; ++mt)
#pragma unroll
      for (int nn = 0; nn < 2; ++nn) {
        int tIdx = mt * 8 + wg * 2 + nn;
        int m = (wg * 2 + nn) * 16 + col;
        float pb2 = pbias[m];
#pragma unroll
        for (int r = 0; r < 8; ++r) {
          int t = mt * 16 + r + half * 8;
          float val = acc2[mt][nn][r] + pacc[tIdx * 256 + (r + half * 8) * 16 + col] + pb2;
          out2T[(m + 6) * 32 + t] = f2bf(val);
        }
      }
  }
  for (int idx = tid; idx < 384; idx += THREADS) {     // zero halo rows of out2T
    int rr = idx >> 5, c = idx & 31;
    int row = (rr < 6) ? rr : (128 + rr);
    out2T[row * 32 + c] = f2bf(0.f);
  }
  for (int idx = tid; idx < 26624; idx += THREADS) {   // fc_w (64,32,13) -> [j][c][t]
    int c = idx / 416, r2 = idx % 416, t = r2 / 13, j = r2 % 13;
    wf[(j * 64 + c) * 32 + t] = f2bf(fc_w[idx]);
  }
  __syncthreads();

  // ---- final conv: M=c(64), N=m(128), K=t(32); 32 tiles, 4 per wave, pipelined ----
  for (int i = 0; i < 4; ++i) {
    int tile = wave * 4 + i;
    int mt = tile >> 3, nt = tile & 7;
    f32x8 acc = {};
    bf16x16 fa = load_frag(wf, mt * 16, 32, 0);
    bf16x16 fb = load_frag(out2T, nt * 16, 32, 0);
#pragma unroll
    for (int j = 0; j < 13; ++j) {
      int nx = (j + 1 < 13) ? j + 1 : j;
      bf16x16 na = load_frag(wf + nx * 2048, mt * 16, 32, 0);
      bf16x16 nb = load_frag(out2T, nt * 16 + nx, 32, 0);
      acc = wmma_bf16(fa, fb, acc);
      fa = na; fb = nb;
    }
#pragma unroll
    for (int r = 0; r < 8; ++r) {
      int c = mt * 16 + r + half * 8;
      zbuf[c * 128 + nt * 16 + col] = acc[r] + fcb[c];
    }
  }
  // warm caches for the residual re-read of q
  for (int off = tid * 16; off < 8192; off += THREADS * 16)
    __builtin_prefetch(xin[0] + off);
  __syncthreads();

  // ---- residual + custom LayerNorm (unbiased std, /(std+eps)) ----
  // 4 threads per row c; quad reduction via lane shuffles.
  {
    int c = tid >> 2, sub = tid & 3;
    const float* qrow = q + ((size_t)b * 64 + c) * 128;
    float* zr = zbuf + c * 128;
    float s = 0.f;
#pragma unroll
    for (int i = 0; i < 32; ++i) {
      int m2 = sub * 32 + i;
      float z = zr[m2] + qrow[m2];
      zr[m2] = z; s += z;
    }
    s += __shfl_xor(s, 1); s += __shfl_xor(s, 2);
    float mu = s * (1.f / 128.f);
    float ss = 0.f;
#pragma unroll
    for (int i = 0; i < 32; ++i) {
      int m2 = sub * 32 + i;
      float d2 = zr[m2] - mu; ss += d2 * d2;
    }
    ss += __shfl_xor(ss, 1); ss += __shfl_xor(ss, 2);
    float sigma = sqrtf(ss * (1.f / 127.f));
    float inv = 1.f / (sigma + 1e-3f);
    float* orow = out + ((size_t)b * 64 + c) * 128;
#pragma unroll
    for (int i = 0; i < 32; ++i) {
      int m2 = sub * 32 + i;
      orow[m2] = (zr[m2] - mu) * inv * lna[m2] + lnb[m2];
    }
  }
}

extern "C" void kernel_launch(void* const* d_in, const int* in_sizes, int n_in,
                              void* d_out, int out_size, void* d_ws, size_t ws_size,
                              hipStream_t stream) {
  (void)in_sizes; (void)n_in; (void)d_ws; (void)ws_size; (void)out_size;
  const float* q      = (const float*)d_in[0];
  const float* k      = (const float*)d_in[1];
  const float* v      = (const float*)d_in[2];
  const float* cq_w   = (const float*)d_in[3];
  const float* cq_b   = (const float*)d_in[4];
  const float* ck_w   = (const float*)d_in[5];
  const float* ck_b   = (const float*)d_in[6];
  const float* cv_w   = (const float*)d_in[7];
  const float* cv_b   = (const float*)d_in[8];
  const float* w_qs   = (const float*)d_in[9];
  const float* w_ks   = (const float*)d_in[10];
  const float* w_vs   = (const float*)d_in[11];
  const float* proj_w = (const float*)d_in[12];
  const float* proj_b = (const float*)d_in[13];
  const float* fc_w   = (const float*)d_in[14];
  const float* fc_b   = (const float*)d_in[15];
  const float* ln_a   = (const float*)d_in[16];
  const float* ln_b   = (const float*)d_in[17];
  float* outp = (float*)d_out;

  hipFuncSetAttribute(reinterpret_cast<const void*>(ha_kernel),
                      hipFuncAttributeMaxDynamicSharedMemorySize, SMEM_BYTES);
  ha_kernel<<<2048, THREADS, SMEM_BYTES, stream>>>(
      q, k, v, cq_w, cq_b, ck_w, ck_b, cv_w, cv_b,
      w_qs, w_ks, w_vs, proj_w, proj_b, fc_w, fc_b, ln_a, ln_b, outp);
}
